// Stage_2_Module_71055938945221
// MI455X (gfx1250) — matrix operand order repaired and checked by reading, not compiled, not run on hardware
//
#include <hip/hip_runtime.h>
#include <hip/hip_bf16.h>

typedef __attribute__((ext_vector_type(16))) _Float16 v16h;
typedef __attribute__((ext_vector_type(8)))  _Float16 v8h;
typedef __attribute__((ext_vector_type(8)))  float    v8f;

#define BB 8
#define HH 128
#define WW 128
#define HW (HH * WW)
#define GROUPS 8

// ---------------------------------------------------------------------------
// Elementwise helpers
// ---------------------------------------------------------------------------

// concat(x,y) [B,64,H,W]+[B,64,H,W] -> f16 [B,128,H,W]
__global__ void pack_concat2_f16(const float* __restrict__ a, const float* __restrict__ b,
                                 _Float16* __restrict__ dst, long total) {
    for (long i = (long)blockIdx.x * blockDim.x + threadIdx.x; i < total;
         i += (long)gridDim.x * blockDim.x) {
        int  bn = (int)(i / (128L * HW));
        long r  = i % (128L * HW);
        int  c  = (int)(r / HW);
        long p  = r % HW;
        float v = (c < 64) ? a[((long)(bn * 64 + c)) * HW + p]
                           : b[((long)(bn * 64 + c - 64)) * HW + p];
        dst[i] = (_Float16)v;
    }
}

// bilinear x2 upsample (align_corners=True) [B,64,64,64] f32 -> f16 channels
// [choff..choff+63] of a [B,cstride,128,128] tensor, scaled by `scale`.
__global__ void up2_f16(const float* __restrict__ src, _Float16* __restrict__ dst,
                        int cstride, int choff, float scale, long total) {
    for (long i = (long)blockIdx.x * blockDim.x + threadIdx.x; i < total;
         i += (long)gridDim.x * blockDim.x) {
        int  bn = (int)(i / (64L * HW));
        long r  = i % (64L * HW);
        int  c  = (int)(r / HW);
        long p  = r % HW;
        int  y  = (int)(p / WW), x = (int)(p % WW);
        float sy = y * (63.0f / 127.0f);
        int   iy = (int)sy; if (iy > 62) iy = 62;
        float fy = sy - (float)iy;
        float sx = x * (63.0f / 127.0f);
        int   ix = (int)sx; if (ix > 62) ix = 62;
        float fx = sx - (float)ix;
        const float* s = src + ((long)(bn * 64 + c)) * 64 * 64;
        float v00 = s[iy * 64 + ix],       v01 = s[iy * 64 + ix + 1];
        float v10 = s[(iy + 1) * 64 + ix], v11 = s[(iy + 1) * 64 + ix + 1];
        float v = (v00 * (1.f - fx) + v01 * fx) * (1.f - fy)
                + (v10 * (1.f - fx) + v11 * fx) * fy;
        dst[((long)(bn * cstride + choff + c)) * HW + p] = (_Float16)(v * scale);
    }
}

// pack conv weights [cout][cin][3][3] f32 -> A [coutPad][9*cin] f16,
// A[co][tap*cin+ci] = w[co][ci][tap]; zero rows for co >= cout.
__global__ void pack_wtap(const float* __restrict__ w, _Float16* __restrict__ A,
                          int cout, int cin, long total) {
    for (long i = (long)blockIdx.x * blockDim.x + threadIdx.x; i < total;
         i += (long)gridDim.x * blockDim.x) {
        int  co  = (int)(i / (9L * cin));
        long r   = i % (9L * cin);
        int  tap = (int)(r / cin);
        int  ci  = (int)(r % cin);
        _Float16 v = (_Float16)0.f;
        if (co < cout) v = (_Float16)w[(((long)co * cin + ci) * 9) + tap];
        A[i] = v;
    }
}

// straight f32 -> f16 convert (DCN weights: native OIHW flatten == G,Cg,K2 flatten)
__global__ void cvt_f16(const float* __restrict__ s, _Float16* __restrict__ d, long total) {
    for (long i = (long)blockIdx.x * blockDim.x + threadIdx.x; i < total;
         i += (long)gridDim.x * blockDim.x)
        d[i] = (_Float16)s[i];
}

// ---------------------------------------------------------------------------
// 3x3 SAME conv as implicit GEMM on WMMA with 2M x 2N register blocking.
// Block = 128 threads (4 waves), owns a 64-pixel row strip x 64 Cout rows.
// Wave w: M-tile pair = blockIdx.z*4 + (w&1)*2, N-tile pair = (w>>1)*2.
// grid = (H*(W/64), B, coutPad/64)
// ---------------------------------------------------------------------------
template <int CIN>
__global__ void __launch_bounds__(128)
conv3x3_wmma(const _Float16* __restrict__ in,   // [B][CIN][H][W]
             const _Float16* __restrict__ wA,   // [coutPad][9*CIN] tap-major K
             const float* __restrict__ bias,    // [cout]
             float* __restrict__ out32,         // nullable [B][cout][H][W]
             _Float16* __restrict__ out16,      // nullable [B][cstride16][H][W]
             int cout, int cstride16, int choff16, int doLrelu) {
    __shared__ _Float16 smem[3 * 66 * CIN];     // halo tile, channel-innermost

    const int tile = blockIdx.x;
    const int y    = tile >> 1;                 // W/64 == 2 tiles per row
    const int x0   = (tile & 1) << 6;
    const int b    = blockIdx.y;
    const int lane = threadIdx.x & 31;
    const int wv   = threadIdx.x >> 5;

    for (int idx = threadIdx.x; idx < 3 * 66 * CIN; idx += 128) {
        int ci  = idx % CIN;
        int p   = idx / CIN;
        int col = p % 66;
        int row = p / 66;
        int yy  = y + row - 1;
        int xx  = x0 + col - 1;
        _Float16 v = (_Float16)0.f;
        if (yy >= 0 && yy < HH && xx >= 0 && xx < WW)
            v = in[((long)(b * CIN + ci)) * HW + (long)yy * WW + xx];
        smem[idx] = v;
    }
    __syncthreads();

    const int  mPair = blockIdx.z * 4 + (wv & 1) * 2;  // co tiles mPair, mPair+1
    const int  nPair = (wv >> 1) * 2;                  // pixel tiles nPair, nPair+1
    const long K     = 9L * CIN;
    // A fragment base: lanes 0-15 hold K 0..15, lanes 16-31 hold 16..31
    const _Float16* arow0 = wA + (long)(mPair * 16 + (lane & 15)) * K + ((lane >> 4) << 4);
    const _Float16* arow1 = arow0 + 16 * K;
    const int nx  = lane & 15;
    const int kk0 = (lane >> 4) << 4;                  // B frag: K 0..15 or 16..31

    v8f acc00 = {}, acc01 = {}, acc10 = {}, acc11 = {};
#pragma unroll
    for (int tap = 0; tap < 9; ++tap) {
        const int ky = tap / 3;
        const int kx = tap - ky * 3;
        const _Float16* brow0 = smem + (ky * 66 + nPair * 16 + nx + kx) * CIN + kk0;
        const _Float16* brow1 = brow0 + 16 * CIN;
#pragma unroll
        for (int cb = 0; cb < CIN; cb += 32) {
            union { v16h v; v8h h[2]; } a0, a1, f0, f1;
            const _Float16* ap0 = arow0 + tap * CIN + cb;
            const _Float16* ap1 = arow1 + tap * CIN + cb;
            a0.h[0] = *(const v8h*)(ap0);
            a0.h[1] = *(const v8h*)(ap0 + 8);
            a1.h[0] = *(const v8h*)(ap1);
            a1.h[1] = *(const v8h*)(ap1 + 8);
            f0.v    = *(const v16h*)(brow0 + cb);
            f1.v    = *(const v16h*)(brow1 + cb);
            acc00 = __builtin_amdgcn_wmma_f32_16x16x32_f16(
                false, a0.v, false, f0.v, (short)0, acc00, false, false);
            acc01 = __builtin_amdgcn_wmma_f32_16x16x32_f16(
                false, a0.v, false, f1.v, (short)0, acc01, false, false);
            acc10 = __builtin_amdgcn_wmma_f32_16x16x32_f16(
                false, a1.v, false, f0.v, (short)0, acc10, false, false);
            acc11 = __builtin_amdgcn_wmma_f32_16x16x32_f16(
                false, a1.v, false, f1.v, (short)0, acc11, false, false);
        }
    }

    const int mb = (lane >> 4) << 3;            // C/D layout: M = j + (lane>=16 ? 8 : 0)
#pragma unroll
    for (int mi = 0; mi < 2; ++mi) {
#pragma unroll
        for (int ni = 0; ni < 2; ++ni) {
            v8f acc = (mi == 0) ? (ni == 0 ? acc00 : acc01)
                                : (ni == 0 ? acc10 : acc11);
            const int xo = x0 + (nPair + ni) * 16 + nx;
#pragma unroll
            for (int j = 0; j < 8; ++j) {
                int co = (mPair + mi) * 16 + mb + j;
                if (co < cout) {
                    float v = acc[j] + bias[co];
                    if (doLrelu) v = (v >= 0.f) ? v : 0.1f * v;
                    long pix = (long)y * WW + xo;
                    if (out32) out32[((long)(b * cout + co)) * HW + pix] = v;
                    if (out16) out16[((long)(b * cstride16 + co + choff16)) * HW + pix] = (_Float16)v;
                }
            }
        }
    }
}

// ---------------------------------------------------------------------------
// Modulated deformable conv (G=8, Cg=8, 3x3) as sampled-im2col + WMMA GEMM.
// Block = 128 threads (4 waves), 16 pixels, Cout=64. grid = (H*W/16, B)
// ---------------------------------------------------------------------------
__global__ void __launch_bounds__(128)
dcn_wmma(const float* __restrict__ xin,       // [B][64][H][W] f32
         const _Float16* __restrict__ oo,     // [B][216][H][W] f16 (dy|dx|mask-logit)
         const _Float16* __restrict__ wA,     // [64][576] f16
         const float* __restrict__ bias,      // [64]
         _Float16* __restrict__ out16) {      // [B][128][H][W], write ch 0..63
    __shared__ _Float16 vals[16 * 576];       // [pixel][K] — K contiguous

    const int tile = blockIdx.x;
    const int y    = tile >> 3;
    const int x0   = (tile & 7) << 4;
    const int b    = blockIdx.y;
    const int lane = threadIdx.x & 31;
    const int wv   = threadIdx.x >> 5;

    for (int t = threadIdx.x; t < 16 * GROUPS * 9; t += 128) {
        int k = t % 9;
        int g = (t / 9) % GROUPS;
        int n = t / (9 * GROUPS);
        int x = x0 + n;
        long opix  = (long)y * WW + x;
        long obase = (long)b * 216 * HW;
        int  chk   = g * 9 + k;
        float dy = (float)oo[obase + (long)chk * HW + opix];
        float dx = (float)oo[obase + (long)(72 + chk) * HW + opix];
        float ml = (float)oo[obase + (long)(144 + chk) * HW + opix];
        float m  = 1.f / (1.f + __expf(-ml));
        float py = (float)y + (float)(k / 3 - 1) + dy;
        float px = (float)x + (float)(k % 3 - 1) + dx;
        float y0f = floorf(py), x0f = floorf(px);
        float fy = py - y0f, fx = px - x0f;
        int iy0 = (int)y0f, ix0 = (int)x0f;
        float w00 = (1.f - fy) * (1.f - fx), w01 = (1.f - fy) * fx;
        float w10 = fy * (1.f - fx),         w11 = fy * fx;
        int y0c = iy0 < 0 ? 0 : (iy0 > HH - 1 ? HH - 1 : iy0);
        int y1c = iy0 + 1 < 0 ? 0 : (iy0 + 1 > HH - 1 ? HH - 1 : iy0 + 1);
        int x0c = ix0 < 0 ? 0 : (ix0 > WW - 1 ? WW - 1 : ix0);
        int x1c = ix0 + 1 < 0 ? 0 : (ix0 + 1 > WW - 1 ? WW - 1 : ix0 + 1);
        float okY0 = (iy0 >= 0 && iy0 < HH) ? 1.f : 0.f;
        float okY1 = (iy0 + 1 >= 0 && iy0 + 1 < HH) ? 1.f : 0.f;
        float okX0 = (ix0 >= 0 && ix0 < WW) ? 1.f : 0.f;
        float okX1 = (ix0 + 1 >= 0 && ix0 + 1 < WW) ? 1.f : 0.f;
        w00 *= okY0 * okX0; w01 *= okY0 * okX1;
        w10 *= okY1 * okX0; w11 *= okY1 * okX1;
#pragma unroll
        for (int c = 0; c < 8; ++c) {
            const float* pl = xin + ((long)(b * 64 + g * 8 + c)) * HW;
            float s = w00 * pl[y0c * WW + x0c] + w01 * pl[y0c * WW + x1c]
                    + w10 * pl[y1c * WW + x0c] + w11 * pl[y1c * WW + x1c];
            vals[n * 576 + (g * 8 + c) * 9 + k] = (_Float16)(s * m);
        }
    }
    __syncthreads();

    const int coTile = wv;                         // Cout = 64 -> 4 tiles
    const _Float16* arow = wA + (long)(coTile * 16 + (lane & 15)) * 576 + ((lane >> 4) << 4);
    const int nx  = lane & 15;
    const int kk0 = (lane >> 4) << 4;
    const _Float16* brow = vals + nx * 576 + kk0;

    v8f acc = {};
#pragma unroll
    for (int cb = 0; cb < 576; cb += 32) {
        union { v16h v; v8h h[2]; } afr, bfr;
        afr.h[0] = *(const v8h*)(arow + cb);
        afr.h[1] = *(const v8h*)(arow + cb + 8);
        bfr.v    = *(const v16h*)(brow + cb);
        acc = __builtin_amdgcn_wmma_f32_16x16x32_f16(
            false, afr.v, false, bfr.v, (short)0, acc, false, false);
    }

    const int mb = (lane >> 4) << 3;
    const int xo = x0 + nx;
#pragma unroll
    for (int j = 0; j < 8; ++j) {
        int co = coTile * 16 + mb + j;
        float v = acc[j] + bias[co];
        out16[((long)(b * 128 + co)) * HW + (long)y * WW + xo] = (_Float16)v;
    }
}

// ---------------------------------------------------------------------------

static inline int ewBlocks(long total) {
    long g = (total + 255) / 256;
    if (g > 16384) g = 16384;
    return (int)g;
}

extern "C" void kernel_launch(void* const* d_in, const int* in_sizes, int n_in,
                              void* d_out, int out_size, void* d_ws, size_t ws_size,
                              hipStream_t stream) {
    (void)in_sizes; (void)n_in; (void)out_size; (void)ws_size;
    const float* x   = (const float*)d_in[0];
    const float* yv  = (const float*)d_in[1];
    const float* po  = (const float*)d_in[2];
    const float* pf  = (const float*)d_in[3];
    const float* w1  = (const float*)d_in[4];  const float* b1  = (const float*)d_in[5];
    const float* w2  = (const float*)d_in[6];  const float* b2  = (const float*)d_in[7];
    const float* w3  = (const float*)d_in[8];  const float* b3  = (const float*)d_in[9];
    const float* wo  = (const float*)d_in[10]; const float* bo  = (const float*)d_in[11];
    const float* wd  = (const float*)d_in[12]; const float* bd  = (const float*)d_in[13];
    const float* wf  = (const float*)d_in[14]; const float* bfb = (const float*)d_in[15];

    char*  ws  = (char*)d_ws;
    size_t pos = 0;
    auto alloc = [&](size_t bytes) -> char* {
        char* p = ws + pos;
        pos += (bytes + 255) & ~(size_t)255;
        return p;
    };
    const long S = HW;
    _Float16* in1  = (_Float16*)alloc((size_t)BB * 128 * S * 2);  // concat(x,y)
    _Float16* in2  = (_Float16*)alloc((size_t)BB * 128 * S * 2);  // concat(off1, 2*up2(po))
    _Float16* off2 = (_Float16*)alloc((size_t)BB * 64  * S * 2);
    _Float16* off3 = (_Float16*)alloc((size_t)BB * 64  * S * 2);
    _Float16* oob  = (_Float16*)alloc((size_t)BB * 216 * S * 2);  // dy|dx|mask logits
    _Float16* finp = (_Float16*)alloc((size_t)BB * 128 * S * 2);  // concat(dcn, up2(pf))
    _Float16* w1p  = (_Float16*)alloc((size_t)64  * 9 * 128 * 2);
    _Float16* w2p  = (_Float16*)alloc((size_t)64  * 9 * 128 * 2);
    _Float16* w3p  = (_Float16*)alloc((size_t)64  * 9 * 64  * 2);
    _Float16* wop  = (_Float16*)alloc((size_t)256 * 9 * 64  * 2); // padded to 256 rows
    _Float16* wfp  = (_Float16*)alloc((size_t)64  * 9 * 128 * 2);
    _Float16* wdp  = (_Float16*)alloc((size_t)64  * 576 * 2);

    float* out_off = (float*)d_out;
    float* out_fea = out_off + (long)BB * 64 * S;

    // --- prep ---
    {
        long t = (long)BB * 128 * S;
        pack_concat2_f16<<<ewBlocks(t), 256, 0, stream>>>(x, yv, in1, t);
    }
    {
        long t = (long)BB * 64 * S;
        up2_f16<<<ewBlocks(t), 256, 0, stream>>>(po, in2, 128, 64, 2.0f, t);
        up2_f16<<<ewBlocks(t), 256, 0, stream>>>(pf, finp, 128, 64, 1.0f, t);
    }
    {
        long t;
        t = 64L  * 9 * 128; pack_wtap<<<ewBlocks(t), 256, 0, stream>>>(w1, w1p, 64,  128, t);
        t = 64L  * 9 * 128; pack_wtap<<<ewBlocks(t), 256, 0, stream>>>(w2, w2p, 64,  128, t);
        t = 64L  * 9 * 64;  pack_wtap<<<ewBlocks(t), 256, 0, stream>>>(w3, w3p, 64,  64,  t);
        t = 256L * 9 * 64;  pack_wtap<<<ewBlocks(t), 256, 0, stream>>>(wo, wop, 216, 64,  t);
        t = 64L  * 9 * 128; pack_wtap<<<ewBlocks(t), 256, 0, stream>>>(wf, wfp, 64,  128, t);
        t = 64L  * 576;     cvt_f16  <<<ewBlocks(t), 256, 0, stream>>>(wd, wdp, t);
    }

    // --- conv chain (64-pixel strips: grid.x = H * W/64 = 256) ---
    dim3 cgrid(HH * (WW / 64), BB, 1);   // 256 x 8
    dim3 cblk(128);
    // off1 = lrelu(conv(in1,w1,b1)) -> ch 0..63 of in2
    conv3x3_wmma<128><<<cgrid, cblk, 0, stream>>>(in1, w1p, b1, nullptr, in2, 64, 128, 0, 1);
    // off2 = lrelu(conv(in2,w2,b2))
    conv3x3_wmma<128><<<cgrid, cblk, 0, stream>>>(in2, w2p, b2, nullptr, off2, 64, 64, 0, 1);
    // off3 = lrelu(conv(off2,w3,b3)) -> output[0] (f32) + f16 copy
    conv3x3_wmma<64><<<cgrid, cblk, 0, stream>>>(off2, w3p, b3, out_off, off3, 64, 64, 0, 1);
    // oo = conv(off3,wo,bo), 216 channels (padded 256 rows -> grid.z = 4)
    dim3 ogrid(HH * (WW / 64), BB, 4);
    conv3x3_wmma<64><<<ogrid, cblk, 0, stream>>>(off3, wop, bo, nullptr, oob, 216, 216, 0, 0);
    // fea0 = dcn(x, oo, wd, bd) -> ch 0..63 of finp
    dim3 dgrid(HH * (WW / 16), BB, 1);   // 1024 x 8
    dcn_wmma<<<dgrid, cblk, 0, stream>>>(x, oob, wdp, bd, finp);
    // fea = lrelu(conv(finp,wf,bf)) -> output[1] (f32)
    conv3x3_wmma<128><<<cgrid, cblk, 0, stream>>>(finp, wfp, bfb, out_fea, nullptr, 64, 0, 0, 1);
}
